// RoPEAttention_17660905521523
// MI455X (gfx1250) — matrix operand order repaired and checked by reading, not compiled, not run on hardware
//
#include <hip/hip_runtime.h>

typedef __attribute__((ext_vector_type(16))) __bf16 v16bf;
typedef __attribute__((ext_vector_type(8)))  __bf16 v8bf;
typedef __attribute__((ext_vector_type(4)))  __bf16 v4bf;
typedef __attribute__((ext_vector_type(8)))  float  v8f;
typedef __attribute__((ext_vector_type(4)))  float  v4f;

#define WMMA_BF16(A, B, Cacc) \
  __builtin_amdgcn_wmma_f32_16x16x32_bf16(false, (A), false, (B), (short)0, (Cacc), false, false)

namespace {
constexpr int kB   = 16;
constexpr int kN   = 1025;
constexpr int kC   = 768;
constexpr int kH   = 12;
constexpr int kHD  = 64;
constexpr int kRows = kB * kN;     // 16400
constexpr int kQKVC = 3 * kC;      // 2304
constexpr int kVStr = 1040;        // padded token stride for vT (32B-aligned rows)
}

// ---------------------------------------------------------------------------
// Kernel 0a: x fp32 -> bf16 (coalesced, 4 elems/thread). 12,595,200 % 4 == 0.
// ---------------------------------------------------------------------------
__global__ void rope_attn_xcvt(const float* __restrict__ x, __bf16* __restrict__ xb, long n) {
    long i = ((long)blockIdx.x * 256 + threadIdx.x) * 4;
    if (i >= n) return;
    v4f v = *(const v4f*)(x + i);
    v4bf o;
#pragma unroll
    for (int j = 0; j < 4; ++j) o[j] = (__bf16)v[j];
    *(v4bf*)(xb + i) = o;
}

// ---------------------------------------------------------------------------
// Kernel 0b: weights fp32 W[K][Nout] -> bf16 WT[Nout][K] (transposed) so WMMA
// B-fragments (16 contiguous K values of one output column) are 32B loads.
// ---------------------------------------------------------------------------
__global__ void rope_attn_wcvt(const float* __restrict__ W, __bf16* __restrict__ WT,
                               int K, int Nout) {
    long i = (long)blockIdx.x * 256 + threadIdx.x;   // i = n*K + k
    if (i >= (long)K * Nout) return;
    int k = (int)(i % K);
    int n = (int)(i / K);
    WT[i] = (__bf16)W[(long)k * Nout + n];
}

// ---------------------------------------------------------------------------
// Kernel 1: qkv = x @ W_qkv, 64x64 register-blocked per wave (4x4 fragments,
// 16 WMMAs per k-step), fused RoPE + scale-fold + scatter:
//   q,k -> [B,H,N,64] bf16 ; v -> transposed [B,H,64,kVStr] bf16.
// A rows clamped (keeps EXEC full for WMMA); stores predicated.
// ---------------------------------------------------------------------------
__global__ void __launch_bounds__(32)
rope_attn_qkv(const __bf16* __restrict__ xb, const __bf16* __restrict__ wT,
              const float* __restrict__ cosT, const float* __restrict__ sinT,
              __bf16* __restrict__ qO, __bf16* __restrict__ kO, __bf16* __restrict__ vT) {
    const int lane = threadIdx.x;
    const int nlo  = lane & 15;
    const int half = lane >> 4;
    const int rowBase = blockIdx.y * 64;
    const int colBase = blockIdx.x * 64;

    const __bf16* aB[4];
#pragma unroll
    for (int mf = 0; mf < 4; ++mf) {
        int r = rowBase + mf * 16 + nlo;
        if (r > kRows - 1) r = kRows - 1;          // clamp ragged M (stores guarded)
        aB[mf] = xb + (long)r * kC;
    }
    const __bf16* bB[4];
#pragma unroll
    for (int nf = 0; nf < 4; ++nf)
        bB[nf] = wT + (long)(colBase + nf * 16 + nlo) * kC;

    v8f acc[4][4] = {};
    for (int k0 = 0; k0 < kC; k0 += 32) {
        v16bf A[4];
#pragma unroll
        for (int mf = 0; mf < 4; ++mf) {
            v8bf lo = *(const v8bf*)(aB[mf] + k0 + half * 16);
            v8bf hi = *(const v8bf*)(aB[mf] + k0 + half * 16 + 8);
#pragma unroll
            for (int j = 0; j < 8; ++j) { A[mf][j] = lo[j]; A[mf][8 + j] = hi[j]; }
        }
#pragma unroll
        for (int nf = 0; nf < 4; ++nf) {
            v16bf Bv = *(const v16bf*)(bB[nf] + k0 + half * 16);
#pragma unroll
            for (int mf = 0; mf < 4; ++mf)
                acc[nf][mf] = WMMA_BF16(A[mf], Bv, acc[nf][mf]);
        }
    }

    // Epilogue: C layout (m,n): lane = n + 16*(m>=8), vgpr r = m%8.
#pragma unroll
    for (int nf = 0; nf < 4; ++nf) {
        const int ncol   = colBase + nf * 16 + nlo;
        const int qkvIdx = ncol / kC;          // uniform per 16-lane group (64|768)
        const int rem    = ncol - qkvIdx * kC;
        const int h      = rem >> 6;           // uniform per group
        const int d      = rem & 63;
        const int fi     = d >> 1;
#pragma unroll
        for (int mf = 0; mf < 4; ++mf) {
#pragma unroll
            for (int r = 0; r < 8; ++r) {
                const int row = rowBase + mf * 16 + r + 8 * half;
                const float val     = acc[nf][mf][r];
                const float partner = __shfl_xor(val, 1, 32);  // before divergence
                if (row >= kRows) continue;
                const int bb  = row / kN;
                const int tok = row - bb * kN;
                if (qkvIdx < 2) {
                    float o = val;
                    if (tok >= 1) {
                        const float cv = cosT[(long)(tok - 1) * (kHD / 2) + fi];
                        const float sv = sinT[(long)(tok - 1) * (kHD / 2) + fi];
                        o = (d & 1) ? (partner * sv + val * cv)
                                    : (val * cv - partner * sv);
                    }
                    const long idx = ((long)(bb * kH + h) * kN + tok) * kHD + d;
                    if (qkvIdx == 0) qO[idx] = (__bf16)(o * 0.125f);   // fold hd^-0.5
                    else             kO[idx] = (__bf16)o;
                } else {
                    vT[((long)(bb * kH + h) * kHD + d) * kVStr + tok] = (__bf16)val;
                }
            }
        }
    }
}

// ---------------------------------------------------------------------------
// Kernel 2: flash attention, TWO 16-query tiles per wave sharing each 32-key
// K/V block (halves K/V traffic, 16 WMMAs per block). Online softmax in fp32,
// P staged through LDS (C-layout -> A-layout).
// ---------------------------------------------------------------------------
__global__ void __launch_bounds__(32)
rope_attn_flash(const __bf16* __restrict__ q, const __bf16* __restrict__ k,
                const __bf16* __restrict__ vT, __bf16* __restrict__ out) {
    __shared__ __attribute__((aligned(32))) __bf16 pl[2][16][32];

    const int lane = threadIdx.x;
    const int nlo  = lane & 15;
    const int half = lane >> 4;
    const int qp = blockIdx.x;     // 0..32 : pair of query tiles
    const int h  = blockIdx.y;
    const int b  = blockIdx.z;
    const long headQK = ((long)b * kH + h) * kN * kHD;
    const long headV  = ((long)b * kH + h) * kHD * kVStr;

    // Preload Q A-fragments for both tiles (d-chunks 0..31, 32..63); clamp rows.
    v16bf aq[2][2];
#pragma unroll
    for (int qi = 0; qi < 2; ++qi) {
        int qtok = (qp * 2 + qi) * 16 + nlo;
        if (qtok > kN - 1) qtok = kN - 1;
        const __bf16* qrow = q + headQK + (long)qtok * kHD;
#pragma unroll
        for (int c = 0; c < 2; ++c) {
            v8bf lo = *(const v8bf*)(qrow + c * 32 + half * 16);
            v8bf hi = *(const v8bf*)(qrow + c * 32 + half * 16 + 8);
#pragma unroll
            for (int j = 0; j < 8; ++j) { aq[qi][c][j] = lo[j]; aq[qi][c][8 + j] = hi[j]; }
        }
    }

    float Mr[2][8], Lr[2][8];
    v8f O[2][4] = {};
#pragma unroll
    for (int qi = 0; qi < 2; ++qi)
#pragma unroll
        for (int r = 0; r < 8; ++r) { Mr[qi][r] = -1e30f; Lr[qi][r] = 0.f; }

    for (int kt = 0; kt < 33; ++kt) {          // 33 * 32 keys >= 1025
        const int kbase = kt * 32;
        const int t0 = kbase + nlo;
        const int t1 = kbase + 16 + nlo;
        const int t0c = t0 > kN - 1 ? kN - 1 : t0;
        const int t1c = t1 > kN - 1 ? kN - 1 : t1;
        const bool inv0 = t0 > kN - 1;
        const bool inv1 = t1 > kN - 1;
        const __bf16* kr0 = k + headQK + (long)t0c * kHD;
        const __bf16* kr1 = k + headQK + (long)t1c * kHD;
        v16bf b00 = *(const v16bf*)(kr0 + half * 16);
        v16bf b01 = *(const v16bf*)(kr1 + half * 16);
        v16bf b10 = *(const v16bf*)(kr0 + 32 + half * 16);
        v16bf b11 = *(const v16bf*)(kr1 + 32 + half * 16);

        __syncthreads();   // WAR vs previous iteration's LDS reads
#pragma unroll
        for (int qi = 0; qi < 2; ++qi) {
            v8f S0 = {}, S1 = {};
            S0 = WMMA_BF16(aq[qi][0], b00, S0);
            S0 = WMMA_BF16(aq[qi][1], b10, S0);
            S1 = WMMA_BF16(aq[qi][0], b01, S1);
            S1 = WMMA_BF16(aq[qi][1], b11, S1);

            float alpha[8];
#pragma unroll
            for (int r = 0; r < 8; ++r) {
                float s0 = inv0 ? -1e30f : S0[r];
                float s1 = inv1 ? -1e30f : S1[r];
                float mx = fmaxf(s0, s1);                 // row max over 16-lane group
                mx = fmaxf(mx, __shfl_xor(mx, 1, 32));
                mx = fmaxf(mx, __shfl_xor(mx, 2, 32));
                mx = fmaxf(mx, __shfl_xor(mx, 4, 32));
                mx = fmaxf(mx, __shfl_xor(mx, 8, 32));
                const float mn = fmaxf(Mr[qi][r], mx);
                alpha[r] = __expf(Mr[qi][r] - mn);
                Mr[qi][r] = mn;
                const float p0 = __expf(s0 - mn);
                const float p1 = __expf(s1 - mn);
                Lr[qi][r] = Lr[qi][r] * alpha[r] + p0 + p1;
                S0[r] = p0; S1[r] = p1;
            }
#pragma unroll
            for (int t = 0; t < 4; ++t)
#pragma unroll
                for (int r = 0; r < 8; ++r) O[qi][t][r] *= alpha[r];
#pragma unroll
            for (int r = 0; r < 8; ++r) {
                const int m = r + 8 * half;
                pl[qi][m][nlo]      = (__bf16)S0[r];
                pl[qi][m][16 + nlo] = (__bf16)S1[r];
            }
        }
        __syncthreads();   // RAW: staging visible

        v16bf ap[2];
#pragma unroll
        for (int qi = 0; qi < 2; ++qi) {
            const __bf16* pr = &pl[qi][nlo][0];
            v8bf lo = *(const v8bf*)(pr + half * 16);
            v8bf hi = *(const v8bf*)(pr + half * 16 + 8);
#pragma unroll
            for (int j = 0; j < 8; ++j) { ap[qi][j] = lo[j]; ap[qi][8 + j] = hi[j]; }
        }

        int vs = kbase + half * 16;
        if (vs > kN - 1) vs = kN - 1;      // clamped span is fully masked (P==0)
        const __bf16* vb = vT + headV + vs;
#pragma unroll
        for (int t = 0; t < 4; ++t) {      // V frags shared by both query tiles
            v16bf bv = *(const v16bf*)(vb + (long)(t * 16 + nlo) * kVStr);
            O[0][t] = WMMA_BF16(ap[0], bv, O[0][t]);
            O[1][t] = WMMA_BF16(ap[1], bv, O[1][t]);
        }
    }

#pragma unroll
    for (int qi = 0; qi < 2; ++qi) {
#pragma unroll
        for (int r = 0; r < 8; ++r) {
            float s = Lr[qi][r];
            s += __shfl_xor(s, 1, 32);
            s += __shfl_xor(s, 2, 32);
            s += __shfl_xor(s, 4, 32);
            s += __shfl_xor(s, 8, 32);
            const float inv = 1.0f / s;
            const int tok = (qp * 2 + qi) * 16 + r + 8 * half;
            if (tok < kN) {
                __bf16* orow = out + ((long)b * kN + tok) * kC + h * kHD;
                orow[0 * 16 + nlo] = (__bf16)(O[qi][0][r] * inv);
                orow[1 * 16 + nlo] = (__bf16)(O[qi][1][r] * inv);
                orow[2 * 16 + nlo] = (__bf16)(O[qi][2][r] * inv);
                orow[3 * 16 + nlo] = (__bf16)(O[qi][3][r] * inv);
            }
        }
    }
}

// ---------------------------------------------------------------------------
// Kernel 3: out = attn @ W_proj + b_proj, 64x64 register-blocked, fp32 out.
// ---------------------------------------------------------------------------
__global__ void __launch_bounds__(32)
rope_attn_proj(const __bf16* __restrict__ a, const __bf16* __restrict__ wT,
               const float* __restrict__ bias, float* __restrict__ out) {
    const int lane = threadIdx.x;
    const int nlo  = lane & 15;
    const int half = lane >> 4;
    const int rowBase = blockIdx.y * 64;
    const int colBase = blockIdx.x * 64;

    const __bf16* aB[4];
#pragma unroll
    for (int mf = 0; mf < 4; ++mf) {
        int r = rowBase + mf * 16 + nlo;
        if (r > kRows - 1) r = kRows - 1;
        aB[mf] = a + (long)r * kC;
    }
    const __bf16* bB[4];
#pragma unroll
    for (int nf = 0; nf < 4; ++nf)
        bB[nf] = wT + (long)(colBase + nf * 16 + nlo) * kC;

    v8f acc[4][4] = {};
    for (int k0 = 0; k0 < kC; k0 += 32) {
        v16bf A[4];
#pragma unroll
        for (int mf = 0; mf < 4; ++mf) {
            v8bf lo = *(const v8bf*)(aB[mf] + k0 + half * 16);
            v8bf hi = *(const v8bf*)(aB[mf] + k0 + half * 16 + 8);
#pragma unroll
            for (int j = 0; j < 8; ++j) { A[mf][j] = lo[j]; A[mf][8 + j] = hi[j]; }
        }
#pragma unroll
        for (int nf = 0; nf < 4; ++nf) {
            v16bf Bv = *(const v16bf*)(bB[nf] + k0 + half * 16);
#pragma unroll
            for (int mf = 0; mf < 4; ++mf)
                acc[nf][mf] = WMMA_BF16(A[mf], Bv, acc[nf][mf]);
        }
    }
#pragma unroll
    for (int nf = 0; nf < 4; ++nf) {
        const int col  = colBase + nf * 16 + nlo;
        const float bc = bias[col];
#pragma unroll
        for (int mf = 0; mf < 4; ++mf) {
#pragma unroll
            for (int r = 0; r < 8; ++r) {
                const long row = rowBase + mf * 16 + r + 8 * half;
                if (row < kRows) out[row * (long)kC + col] = acc[nf][mf][r] + bc;
            }
        }
    }
}

// ---------------------------------------------------------------------------
extern "C" void kernel_launch(void* const* d_in, const int* in_sizes, int n_in,
                              void* d_out, int out_size, void* d_ws, size_t ws_size,
                              hipStream_t stream) {
    (void)in_sizes; (void)n_in; (void)out_size; (void)ws_size;
    const float* x     = (const float*)d_in[0];
    const float* fcos  = (const float*)d_in[1];
    const float* fsin  = (const float*)d_in[2];
    const float* Wqkv  = (const float*)d_in[3];
    const float* Wproj = (const float*)d_in[4];
    const float* bproj = (const float*)d_in[5];
    float* out = (float*)d_out;

    // Workspace layout (bytes)
    char* ws = (char*)d_ws;
    const size_t szWqkvT  = (size_t)kQKVC * kC * sizeof(__bf16);
    const size_t szWprojT = (size_t)kC * kC * sizeof(__bf16);
    const size_t szXB     = (size_t)kRows * kC * sizeof(__bf16);
    const size_t szQK     = (size_t)kB * kH * kN * kHD * sizeof(__bf16);
    const size_t szVT     = (size_t)kB * kH * kHD * kVStr * sizeof(__bf16);
    __bf16* wqkvT  = (__bf16*)(ws);
    __bf16* wprojT = (__bf16*)(ws + szWqkvT);
    __bf16* xbuf   = (__bf16*)(ws + szWqkvT + szWprojT);
    __bf16* qB     = (__bf16*)(ws + szWqkvT + szWprojT + szXB);
    __bf16* kB_    = (__bf16*)(ws + szWqkvT + szWprojT + szXB + szQK);
    __bf16* vTB    = (__bf16*)(ws + szWqkvT + szWprojT + szXB + 2 * szQK);
    __bf16* attnO  = (__bf16*)(ws + szWqkvT + szWprojT + szXB + 2 * szQK + szVT);

    // 0) one-time conversions
    {
        long nx = (long)kRows * kC;                          // 12,595,200 (mult of 4)
        rope_attn_xcvt<<<dim3((unsigned)((nx / 4 + 255) / 256)), dim3(256), 0, stream>>>(
            x, xbuf, nx);
        long n1 = (long)kC * kQKVC;
        rope_attn_wcvt<<<dim3((unsigned)((n1 + 255) / 256)), dim3(256), 0, stream>>>(
            Wqkv, wqkvT, kC, kQKVC);
        long n2 = (long)kC * kC;
        rope_attn_wcvt<<<dim3((unsigned)((n2 + 255) / 256)), dim3(256), 0, stream>>>(
            Wproj, wprojT, kC, kC);
    }
    // 1) qkv GEMM + RoPE + scatter   (64x64 tiles: 2304/64 x ceil(16400/64))
    rope_attn_qkv<<<dim3(kQKVC / 64, (kRows + 63) / 64), dim3(32), 0, stream>>>(
        xbuf, wqkvT, fcos, fsin, qB, kB_, vTB);
    // 2) flash attention             (33 q-tile pairs x 12 heads x 16 batch)
    rope_attn_flash<<<dim3(33, kH, kB), dim3(32), 0, stream>>>(qB, kB_, vTB, attnO);
    // 3) projection GEMM + bias      (768/64 x ceil(16400/64))
    rope_attn_proj<<<dim3(kC / 64, (kRows + 63) / 64), dim3(32), 0, stream>>>(
        attnO, wprojT, bproj, out);
}
